// CausalSelfMQA_79096117723523
// MI455X (gfx1250) — matrix operand order repaired and checked by reading, not compiled, not run on hardware
//
#include <hip/hip_runtime.h>
#include <hip/hip_bf16.h>

#define B_  2
#define S_  2048
#define D_  2048
#define H_  16
#define DH_ 128

typedef __bf16 bf16_t;
typedef __bf16 v16bf __attribute__((ext_vector_type(16)));
typedef __bf16 v8bf  __attribute__((ext_vector_type(8)));
typedef float  v8f   __attribute__((ext_vector_type(8)));

__device__ __forceinline__ v8f wmma_bf16(v16bf a, v16bf b, v8f c) {
  // 8 args: (neg_a, A, neg_b, B, c_mod, C, reuse_a, reuse_b)
  return __builtin_amdgcn_wmma_f32_16x16x32_bf16(false, a, false, b, (short)0, c,
                                                 false, false);
}

// A-matrix 16x32 bf16 fragment: lane holds row (lane&15),
// k = kh*16+0..15 contiguous where kh = lane>>4  -> two 16B loads.
__device__ __forceinline__ v16bf load_a_frag(const bf16_t* base, int ld, int lane) {
  const int row = lane & 15;
  const int kh  = lane >> 4;
  const bf16_t* p = base + (size_t)row * ld + kh * 16;
  union { v16bf v; v8bf h[2]; } u;
  u.h[0] = *(const v8bf*)(p);
  u.h[1] = *(const v8bf*)(p + 8);
  return u.v;
}

// B-matrix 32x16 bf16 fragment: lane holds column (lane&15),
// k = kh*16+0..15 contiguous -> one 32B run per lane.
__device__ __forceinline__ v16bf load_b_frag(const bf16_t* base, int ld, int lane) {
  const int n  = lane & 15;
  const int kh = lane >> 4;
  return *(const v16bf*)(base + (size_t)n * ld + kh * 16);
}

__global__ void cvt_f32_bf16(const float* __restrict__ in, bf16_t* __restrict__ out,
                             long n) {
  long i = (long)blockIdx.x * blockDim.x + threadIdx.x;
  const long stride = (long)gridDim.x * blockDim.x;
  for (; i < n; i += stride) out[i] = (bf16_t)in[i];
}

// C[M,N] = A[M,K] (bf16, row-major) @ W[N,K]^T (bf16, row-major) + bias, fp32 out.
// Block = 8 waves in 2(M)x4(N); wave tile = 16M x 64N; block tile = 32M x 256N.
__global__ void __launch_bounds__(256) gemm_bf16(
    const bf16_t* __restrict__ A, const bf16_t* __restrict__ W,
    const float* __restrict__ bias, float* __restrict__ out,
    int M, int N, int K) {
  const int lane = threadIdx.x & 31;
  const int w    = threadIdx.x >> 5;
  const int m0   = blockIdx.x * 32 + (w >> 2) * 16;
  const int n0   = blockIdx.y * 256 + (w & 3) * 64;
  if (m0 >= M || n0 >= N) return;
  const int col = lane & 15;
  const int hi  = lane >> 4;

  v8f acc[4] = {};
  const bf16_t* Abase = A + (size_t)m0 * K;
  for (int k = 0; k < K; k += 32) {
    __builtin_prefetch((const void*)(Abase + (size_t)col * K + k + 256), 0, 1);
    v16bf a = load_a_frag(Abase + k, K, lane);
#pragma unroll
    for (int t = 0; t < 4; ++t) {
      v16bf b = load_b_frag(W + (size_t)(n0 + t * 16) * K + k, K, lane);
      acc[t] = wmma_bf16(a, b, acc[t]);
    }
  }
#pragma unroll
  for (int t = 0; t < 4; ++t) {
    const float bv = bias[n0 + t * 16 + col];
#pragma unroll
    for (int i = 0; i < 8; ++i) {
      const int row = i + hi * 8;  // C layout: vgpr i -> M=i (lanes 0-15) / M=8+i
      out[(size_t)(m0 + row) * N + n0 + t * 16 + col] = acc[t][i] + bv;
    }
  }
}

// RoPE on Q (fp32 in, bf16 out), layout (b,s,h,dh).
__global__ void rope_q(const float* __restrict__ q, bf16_t* __restrict__ qo,
                       long total) {
  long t = (long)blockIdx.x * blockDim.x + threadIdx.x;
  if (t >= total) return;
  const int  i  = (int)(t & 63);
  const long r  = t >> 6;       // (b*S+s)*H + h
  const int  h  = (int)(r % H_);
  const long bs = r / H_;       // b*S + s
  const int  s  = (int)(bs % S_);
  const float* src = q + bs * (size_t)D_ + h * DH_;
  bf16_t*      dst = qo + bs * (size_t)D_ + h * DH_;
  const float inv = __expf(-(float)(2 * i) * (9.210340371976184f / 128.f)); // 10000^(-2i/128)
  float sn, cs; __sincosf((float)s * inv, &sn, &cs);
  const float a0 = src[i], a1 = src[i + 64];
  dst[i]      = (bf16_t)(a0 * cs - a1 * sn);
  dst[i + 64] = (bf16_t)(a1 * cs + a0 * sn);
}

// RoPE on K + transpose of V: kv (b,s,256) fp32 -> k_bf (b,s,128), vT (b,128,s).
__global__ void rope_kv(const float* __restrict__ kv, bf16_t* __restrict__ ko,
                        bf16_t* __restrict__ vt, long total) {
  long t = (long)blockIdx.x * blockDim.x + threadIdx.x;
  if (t >= total) return;
  const int  dh = (int)(t % DH_);
  const long bs = t / DH_;
  const int  s  = (int)(bs % S_);
  const int  b  = (int)(bs / S_);
  const float* row = kv + bs * 256;
  const int  i  = dh & 63;
  const float inv = __expf(-(float)(2 * i) * (9.210340371976184f / 128.f));
  float sn, cs; __sincosf((float)s * inv, &sn, &cs);
  float v;
  if (dh < 64) v = row[dh] * cs - row[dh + 64] * sn;
  else         v = row[dh] * cs + row[dh - 64] * sn;
  ko[bs * DH_ + dh] = (bf16_t)v;
  vt[((size_t)b * DH_ + dh) * S_ + s] = (bf16_t)row[128 + dh];
}

// Flash attention, causal, MQA (1 KV head shared by 16 Q heads).
// One wave owns one 16-row Q tile of one (b,h); streams 32 keys / iteration.
// Scores and O accumulator never leave registers; P transits wave-private LDS
// only for the C-layout -> A-layout relayout feeding the PV WMMA.
__global__ void __launch_bounds__(256) attn_fa(
    const bf16_t* __restrict__ qb, const bf16_t* __restrict__ kb,
    const bf16_t* __restrict__ vtb, bf16_t* __restrict__ attn) {
  __shared__ bf16_t p_lds[8][16][32];  // per-wave 16x32 P tile (1 KB each)

  const int lane = threadIdx.x & 31;
  const int w    = threadIdx.x >> 5;
  const int tile = blockIdx.x * 8 + w;
  const int QT   = S_ / 16;
  const int b    = tile / (H_ * QT);
  const int rem  = tile % (H_ * QT);
  const int h    = rem / QT;
  const int q0   = (rem % QT) * 16;
  const int col  = lane & 15;
  const int hi   = lane >> 4;

  const bf16_t* qbase = qb + (size_t)(b * S_ + q0) * D_ + h * DH_;
  v16bf qa[4];
#pragma unroll
  for (int c = 0; c < 4; ++c) qa[c] = load_a_frag(qbase + c * 32, D_, lane);

  const bf16_t* kbase  = kb  + (size_t)b * S_ * DH_;   // (key, dh) row-major
  const bf16_t* vtbase = vtb + (size_t)b * DH_ * S_;   // (dh, key) row-major

  v8f o[8] = {};
  float m_i[8], l_i[8];
#pragma unroll
  for (int i = 0; i < 8; ++i) { m_i[i] = -1e30f; l_i[i] = 0.f; }

  const float scale = 0.088388347648318447f;  // 1/sqrt(128)
  const int kend = q0 + 16;                   // causal key count for this tile

  for (int j = 0; j < kend; j += 32) {
    // ---- scores S = Q @ K^T for keys [j, j+32) : two 16x16 C tiles ----
    v8f c0 = {}, c1 = {};
    const int key0  = j + col;                    // always < S
    const int key1c = min(j + 16 + col, S_ - 1);  // clamp; masked below anyway
#pragma unroll
    for (int c = 0; c < 4; ++c) {
      v16bf bk0 = *(const v16bf*)(kbase + (size_t)key0  * DH_ + c * 32 + hi * 16);
      c0 = wmma_bf16(qa[c], bk0, c0);
      v16bf bk1 = *(const v16bf*)(kbase + (size_t)key1c * DH_ + c * 32 + hi * 16);
      c1 = wmma_bf16(qa[c], bk1, c1);
    }
    // ---- online softmax over the 16x32 tile (rows replicated per 16-lane group)
    float alpha[8];
#pragma unroll
    for (int i = 0; i < 8; ++i) {
      const int row  = i + hi * 8;
      const int qrow = q0 + row;
      float s0 = c0[i] * scale; if (j + col > qrow)      s0 = -1e30f;
      float s1 = c1[i] * scale; if (j + 16 + col > qrow) s1 = -1e30f;
      float mx = fmaxf(s0, s1);
#pragma unroll
      for (int msk = 1; msk < 16; msk <<= 1) mx = fmaxf(mx, __shfl_xor(mx, msk, 32));
      const float mnew = fmaxf(m_i[i], mx);
      const float al   = __expf(m_i[i] - mnew);
      const float p0   = __expf(s0 - mnew);
      const float p1   = __expf(s1 - mnew);
      float rs = p0 + p1;
#pragma unroll
      for (int msk = 1; msk < 16; msk <<= 1) rs += __shfl_xor(rs, msk, 32);
      l_i[i]   = l_i[i] * al + rs;
      m_i[i]   = mnew;
      alpha[i] = al;
      p_lds[w][row][col]      = (bf16_t)p0;
      p_lds[w][row][col + 16] = (bf16_t)p1;
    }
#pragma unroll
    for (int f = 0; f < 8; ++f)
#pragma unroll
      for (int i = 0; i < 8; ++i) o[f][i] *= alpha[i];

    // P stores -> P loads are same-wave LDS; DS ops are in-order, but fence the
    // compiler and drain DScnt before consuming the relayout.
    asm volatile("s_wait_dscnt 0" ::: "memory");

    // ---- O += P @ V : P is exactly one 16x32 A fragment ----
    v16bf pa = load_a_frag(&p_lds[w][0][0], 32, lane);
#pragma unroll
    for (int f = 0; f < 8; ++f) {
      v16bf bv = *(const v16bf*)(vtbase + (size_t)(f * 16 + col) * S_ + j + hi * 16);
      o[f] = wmma_bf16(pa, bv, o[f]);
    }
  }

  // ---- normalize and write attn (b,s,h,dh) as bf16 for the O-projection GEMM
  float inv_l[8];
#pragma unroll
  for (int i = 0; i < 8; ++i) inv_l[i] = 1.f / l_i[i];
  bf16_t* obase = attn + (size_t)(b * S_ + q0) * D_ + h * DH_;
#pragma unroll
  for (int f = 0; f < 8; ++f)
#pragma unroll
    for (int i = 0; i < 8; ++i) {
      const int row = i + hi * 8;
      obase[(size_t)row * D_ + f * 16 + col] = (bf16_t)(o[f][i] * inv_l[i]);
    }
}

extern "C" void kernel_launch(void* const* d_in, const int* in_sizes, int n_in,
                              void* d_out, int out_size, void* d_ws, size_t ws_size,
                              hipStream_t stream) {
  (void)in_sizes; (void)n_in; (void)out_size; (void)ws_size;
  const float* x      = (const float*)d_in[0];
  const float* qw     = (const float*)d_in[1];
  const float* qbias  = (const float*)d_in[2];
  const float* kvw    = (const float*)d_in[3];
  const float* kvbias = (const float*)d_in[4];
  const float* ow     = (const float*)d_in[5];
  const float* obias  = (const float*)d_in[6];
  float* out = (float*)d_out;

  char* p = (char*)d_ws;
  auto carve = [&](size_t bytes) -> char* {
    char* r = p;
    p += (bytes + 255) & ~(size_t)255;
    return r;
  };
  const size_t BS = (size_t)B_ * S_;
  bf16_t* x_bf   = (bf16_t*)carve(BS * D_ * 2);
  bf16_t* qw_bf  = (bf16_t*)carve((size_t)D_ * D_ * 2);
  bf16_t* kvw_bf = (bf16_t*)carve((size_t)2 * DH_ * D_ * 2);
  bf16_t* ow_bf  = (bf16_t*)carve((size_t)D_ * D_ * 2);
  float*  q_f    = (float*)carve(BS * D_ * 4);
  float*  kv_f   = (float*)carve(BS * 2 * DH_ * 4);
  bf16_t* q_bf   = (bf16_t*)carve(BS * D_ * 2);
  bf16_t* k_bf   = (bf16_t*)carve(BS * DH_ * 2);
  bf16_t* vt_bf  = (bf16_t*)carve(BS * DH_ * 2);
  bf16_t* at_bf  = (bf16_t*)carve(BS * D_ * 2);

  auto cdiv = [](long a, long b) { return (unsigned)((a + b - 1) / b); };
  const dim3 blk(256);

  // fp32 -> bf16 one-time conversions (x + 3 weight matrices)
  cvt_f32_bf16<<<2048, blk, 0, stream>>>(x,   x_bf,   (long)BS * D_);
  cvt_f32_bf16<<<2048, blk, 0, stream>>>(qw,  qw_bf,  (long)D_ * D_);
  cvt_f32_bf16<<<512,  blk, 0, stream>>>(kvw, kvw_bf, (long)2 * DH_ * D_);
  cvt_f32_bf16<<<2048, blk, 0, stream>>>(ow,  ow_bf,  (long)D_ * D_);

  // Q projection: (4096 x 2048) = x @ qw^T + qbias
  gemm_bf16<<<dim3((unsigned)(BS / 32), D_ / 256), blk, 0, stream>>>(
      x_bf, qw_bf, qbias, q_f, (int)BS, D_, D_);
  // KV projection: (4096 x 256)
  gemm_bf16<<<dim3((unsigned)(BS / 32), 1), blk, 0, stream>>>(
      x_bf, kvw_bf, kvbias, kv_f, (int)BS, 2 * DH_, D_);

  // RoPE + V transpose
  rope_q<<<cdiv((long)BS * H_ * 64, 256), blk, 0, stream>>>(q_f, q_bf,
                                                            (long)BS * H_ * 64);
  rope_kv<<<cdiv((long)BS * DH_, 256), blk, 0, stream>>>(kv_f, k_bf, vt_bf,
                                                         (long)BS * DH_);

  // Flash attention: B*H*(S/16) = 4096 wave-tiles, 8 waves/block
  attn_fa<<<(B_ * H_ * (S_ / 16)) / 8, blk, 0, stream>>>(q_bf, k_bf, vt_bf, at_bf);

  // Output projection: (4096 x 2048) = attn @ ow^T + obias, fp32 out
  gemm_bf16<<<dim3((unsigned)(BS / 32), D_ / 256), blk, 0, stream>>>(
      at_bf, ow_bf, obias, out, (int)BS, D_, D_);
}
